// Net_60112362275749
// MI455X (gfx1250) — matrix operand order repaired and checked by reading, not compiled, not run on hardware
//
#include <hip/hip_runtime.h>

// ---------------- problem constants (from reference) ----------------
#define N_NODES   100000
#define N_EDGES   1600000
#define F_IN      512
#define HEADS     8
#define C1        8
#define N_CLASSES 40
#define E2        (N_EDGES + N_NODES)   // edges + self loops
#define NEG_SLOPE 0.2f
#define EPS_F     1e-16f

typedef __attribute__((ext_vector_type(16))) __bf16 bf16x16;
typedef __attribute__((ext_vector_type(8)))  __bf16 bf16x8;
typedef __attribute__((ext_vector_type(8)))  float  f32x8;

// ---------------- utility kernels ----------------
__global__ void fill_f32(float* __restrict__ p, float v, long long n) {
  long long i = (long long)blockIdx.x * blockDim.x + threadIdx.x;
  if (i < n) p[i] = v;
}

__global__ void cvt_f32_bf16(const float* __restrict__ in, __bf16* __restrict__ out,
                             long long n) {
  long long i = (long long)blockIdx.x * blockDim.x + threadIdx.x;
  if (i < n) out[i] = (__bf16)in[i];
}

// Pack B[K,Nc] (f32 row-major) into WMMA-fragment-major bf16:
//   Bp[(((kb/32)*(Nc/16) + tile)*32 + lane)*16 + e]
//     = B[(kb + 16*(lane>>4) + e) * Nc + tile*16 + (lane&15)]
// so each lane's 16-element B fragment is 32 contiguous bytes (2x b128 loads).
__global__ void pack_b_bf16(const float* __restrict__ B, __bf16* __restrict__ Bp,
                            int K, int Nc) {
  long long idx = (long long)blockIdx.x * blockDim.x + threadIdx.x;
  long long total = (long long)K * Nc;
  if (idx >= total) return;
  int e    = (int)(idx & 15);
  int lane = (int)((idx >> 4) & 31);
  long long rest = idx >> 9;             // kblk*(Nc/16) + tile
  int ntiles = Nc >> 4;
  int tile = (int)(rest % ntiles);
  int kblk = (int)(rest / ntiles);
  int k = kblk * 32 + 16 * (lane >> 4) + e;
  int n = tile * 16 + (lane & 15);
  Bp[idx] = (__bf16)B[(size_t)k * Nc + n];
}

// ---------------- WMMA GEMM: C[M,Nc] = A[M,K](bf16) @ Bp(packed bf16), f32 accum --
// One wave -> 16 rows x 64 cols (A fragment reused across 4 WMMA accumulators).
// blockDim = 128 (4 waves). grid.x = Nc/64, grid.y = ceil(ceil(M/16)/4).
__launch_bounds__(128)
__global__ void gemm_bf16_wmma(const __bf16* __restrict__ A,
                               const __bf16* __restrict__ Bp,
                               float* __restrict__ C,
                               int M, int K, int Nc) {
  const int lane  = threadIdx.x & 31;
  const int wave  = threadIdx.x >> 5;
  const int mTile = blockIdx.y * 4 + wave;       // 16-row tile index
  const int n0    = blockIdx.x * 64;             // 64-col strip base
  const int t0    = blockIdx.x * 4;              // first 16-col tile index
  const int mLoc  = lane & 15;
  const int kHalf = lane >> 4;                   // selects contiguous K half: k = kb+16*kHalf+0..15
  const int mRow  = mTile * 16 + mLoc;
  const int mSafe = (mRow < M) ? mRow : 0;       // clamp; OOB rows masked at store
  const __bf16* __restrict__ arow = A + (size_t)mSafe * K;
  const int ntiles = Nc >> 4;

  f32x8 acc[4] = {f32x8{}, f32x8{}, f32x8{}, f32x8{}};

  for (int kb = 0; kb < K; kb += 32) {
    if (kb + 32 < K) __builtin_prefetch(arow + kb + 32, 0, 1);
    // A 16x32 bf16 fragment: lane row = lane&15; elems 0-15 -> K=kb+16*kHalf+0..15
    // (two 16B contiguous loads, matching packed B).
    bf16x8 alo = *(const bf16x8*)(arow + kb + 16 * kHalf);
    bf16x8 ahi = *(const bf16x8*)(arow + kb + 16 * kHalf + 8);
    bf16x16 afrag = __builtin_shufflevector(alo, ahi, 0, 1, 2, 3, 4, 5, 6, 7,
                                            8, 9, 10, 11, 12, 13, 14, 15);

    // packed B base for this k-block
    const __bf16* __restrict__ bblk = Bp + (((size_t)(kb >> 5) * ntiles) << 9);
#pragma unroll
    for (int t = 0; t < 4; ++t) {
      const __bf16* __restrict__ bl = bblk + (((size_t)(t0 + t) * 32 + lane) << 4);
      bf16x8 b0 = *(const bf16x8*)(bl);
      bf16x8 b1 = *(const bf16x8*)(bl + 8);
      bf16x16 bfrag = __builtin_shufflevector(b0, b1, 0, 1, 2, 3, 4, 5, 6, 7,
                                              8, 9, 10, 11, 12, 13, 14, 15);
      acc[t] = __builtin_amdgcn_wmma_f32_16x16x32_bf16(
          /*neg_a=*/false, afrag, /*neg_b=*/false, bfrag,
          /*c_mod=*/(short)0, acc[t], /*reuse_a=*/false, /*reuse_b=*/false);
    }
  }

  // C/D layout: VGPR r -> row r + 8*(lane>>4); col = lane&15
#pragma unroll
  for (int t = 0; t < 4; ++t) {
#pragma unroll
    for (int r = 0; r < 8; ++r) {
      int row = mTile * 16 + r + 8 * kHalf;
      int col = n0 + t * 16 + mLoc;
      if (row < M) C[(size_t)row * Nc + col] = acc[t][r];
    }
  }
}

// ---------------- edge attention: MX score + leaky_relu + segment max -----------
template <int C>
__global__ void edge_alpha(const float* __restrict__ xw,
                           const int* __restrict__ esrc, const int* __restrict__ edst,
                           const float* __restrict__ att_l, const float* __restrict__ att_r,
                           float* __restrict__ alpha, float* __restrict__ amax) {
  long long idx = (long long)blockIdx.x * blockDim.x + threadIdx.x;
  if (idx >= (long long)E2 * HEADS) return;
  int e = (int)(idx >> 3);
  int h = (int)(idx & 7);
  int j = (e < N_EDGES) ? esrc[e] : (e - N_EDGES);   // source
  int i = (e < N_EDGES) ? edst[e] : (e - N_EDGES);   // destination
  const float4* __restrict__ xj = (const float4*)(xw + ((size_t)j * HEADS + h) * C);
  const float4* __restrict__ xi = (const float4*)(xw + ((size_t)i * HEADS + h) * C);
  const float4* __restrict__ al = (const float4*)(att_l + h * C);
  const float4* __restrict__ ar = (const float4*)(att_r + h * C);
  float logit = 0.f, sl = 0.f, sr = 0.f;
#pragma unroll
  for (int c = 0; c < C / 4; ++c) {
    float4 a = xj[c], b = xi[c], l4 = al[c], r4 = ar[c];
    logit += a.x * b.x + a.y * b.y + a.z * b.z + a.w * b.w;
    sl    += a.x * l4.x + a.y * l4.y + a.z * l4.z + a.w * l4.w;
    sr    += b.x * r4.x + b.y * r4.y + b.z * r4.z + b.w * r4.w;
  }
  float v = (sl + sr) * (1.f / (1.f + __expf(-logit)));   // GO gated by sigmoid(DP)
  v = (v > 0.f) ? v : NEG_SLOPE * v;                      // leaky relu
  alpha[idx] = v;
  float* mp = amax + (size_t)i * HEADS + h;               // segment max (atomic trick)
  if (v >= 0.f) atomicMax((int*)mp, __float_as_int(v));
  else          atomicMin((unsigned int*)mp, __float_as_uint(v));
}

// ---------------- exp(alpha - max) and segment-sum of denominators --------------
__global__ void edge_expsum(const int* __restrict__ edst,
                            float* __restrict__ alpha,
                            const float* __restrict__ amax,
                            float* __restrict__ denom) {
  long long idx = (long long)blockIdx.x * blockDim.x + threadIdx.x;
  if (idx >= (long long)E2 * HEADS) return;
  int e = (int)(idx >> 3);
  int h = (int)(idx & 7);
  int i = (e < N_EDGES) ? edst[e] : (e - N_EDGES);
  float ex = __expf(alpha[idx] - amax[(size_t)i * HEADS + h]);
  alpha[idx] = ex;
  atomicAdd(denom + (size_t)i * HEADS + h, ex);
}

// ---------------- weighted segment-sum aggregation (4 channels / thread) --------
template <int C>
__global__ void edge_agg(const float* __restrict__ xw,
                         const int* __restrict__ esrc, const int* __restrict__ edst,
                         const float* __restrict__ alpha, const float* __restrict__ denom,
                         float* __restrict__ agg) {
  const int QC = C / 4;                      // float4 groups per head
  long long idx = (long long)blockIdx.x * blockDim.x + threadIdx.x;
  if (idx >= (long long)E2 * HEADS * QC) return;
  int hq = (int)(idx % (HEADS * QC));        // head * QC + quad
  int e  = (int)(idx / (HEADS * QC));
  int h  = hq / QC;
  int q  = hq % QC;
  int j = (e < N_EDGES) ? esrc[e] : (e - N_EDGES);
  int i = (e < N_EDGES) ? edst[e] : (e - N_EDGES);
  float a = alpha[(size_t)e * HEADS + h] / (denom[(size_t)i * HEADS + h] + EPS_F);
  float4 xv = *(const float4*)(xw + ((size_t)j * HEADS + h) * C + q * 4);
  float* dst = agg + ((size_t)i * HEADS + h) * C + q * 4;
  atomicAdd(dst + 0, xv.x * a);
  atomicAdd(dst + 1, xv.y * a);
  atomicAdd(dst + 2, xv.z * a);
  atomicAdd(dst + 3, xv.w * a);
}

// ---------------- layer1 finalize: +bias, ELU, cast to bf16 for GEMM2 -----------
__global__ void finalize1(const float* __restrict__ agg, const float* __restrict__ b1,
                          __bf16* __restrict__ hbf) {
  long long idx = (long long)blockIdx.x * blockDim.x + threadIdx.x;
  if (idx >= (long long)N_NODES * (HEADS * C1)) return;
  int f = (int)(idx & (HEADS * C1 - 1));
  float v = agg[idx] + b1[f];
  v = (v > 0.f) ? v : (__expf(v) - 1.f);   // elu
  hbf[idx] = (__bf16)v;
}

// ---------------- layer2 finalize: head-mean, +bias, log_softmax ---------------
__global__ void finalize2(const float* __restrict__ agg2, const float* __restrict__ b2,
                          float* __restrict__ out) {
  int n = blockIdx.x * blockDim.x + threadIdx.x;
  if (n >= N_NODES) return;
  float vals[N_CLASSES];
  float mx = -__builtin_inff();
#pragma unroll 4
  for (int c = 0; c < N_CLASSES; ++c) {
    float s = 0.f;
#pragma unroll
    for (int h = 0; h < HEADS; ++h)
      s += agg2[((size_t)n * HEADS + h) * N_CLASSES + c];
    float v = s * (1.f / HEADS) + b2[c];
    vals[c] = v;
    mx = fmaxf(mx, v);
  }
  float se = 0.f;
#pragma unroll 4
  for (int c = 0; c < N_CLASSES; ++c) se += __expf(vals[c] - mx);
  float lse = mx + __logf(se);
#pragma unroll 4
  for (int c = 0; c < N_CLASSES; ++c)
    out[(size_t)n * N_CLASSES + c] = vals[c] - lse;
  if (n == 0) out[(size_t)N_NODES * N_CLASSES] = 0.f;   // att_loss (eval mode)
}

// ---------------- launcher ----------------
extern "C" void kernel_launch(void* const* d_in, const int* in_sizes, int n_in,
                              void* d_out, int out_size, void* d_ws, size_t ws_size,
                              hipStream_t stream) {
  const float* x     = (const float*)d_in[0];
  const int*   ei    = (const int*)d_in[1];
  const float* W1    = (const float*)d_in[2];
  const float* attl1 = (const float*)d_in[3];
  const float* attr1 = (const float*)d_in[4];
  const float* b1    = (const float*)d_in[5];
  const float* W2    = (const float*)d_in[6];
  const float* attl2 = (const float*)d_in[7];
  const float* attr2 = (const float*)d_in[8];
  const float* b2    = (const float*)d_in[9];
  const int* esrc = ei;
  const int* edst = ei + N_EDGES;
  float* out = (float*)d_out;

  // bump allocator over workspace (256B aligned chunks)
  char* wp = (char*)d_ws;
  auto alloc = [&](size_t bytes) -> void* {
    void* p = (void*)wp;
    wp += (bytes + 255) & ~(size_t)255;
    return p;
  };
  __bf16* xbf  = (__bf16*)alloc((size_t)N_NODES * F_IN * 2);
  __bf16* w1p  = (__bf16*)alloc((size_t)F_IN * 64 * 2);          // packed W1
  float*  xw1  = (float*) alloc((size_t)N_NODES * HEADS * C1 * 4);
  float*  alp  = (float*) alloc((size_t)E2 * HEADS * 4);
  float*  amax = (float*) alloc((size_t)N_NODES * HEADS * 4);
  float*  den  = (float*) alloc((size_t)N_NODES * HEADS * 4);
  float*  agg1 = (float*) alloc((size_t)N_NODES * HEADS * C1 * 4);
  __bf16* hbf  = (__bf16*)alloc((size_t)N_NODES * HEADS * C1 * 2);
  __bf16* w2p  = (__bf16*)alloc((size_t)64 * 320 * 2);           // packed W2
  float*  xw2  = (float*) alloc((size_t)N_NODES * HEADS * N_CLASSES * 4);
  float*  agg2 = (float*) alloc((size_t)N_NODES * HEADS * N_CLASSES * 4);

  const int TB = 256;
  auto nb = [](long long n, int tb) { return (unsigned)((n + tb - 1) / tb); };
  const unsigned mBlocks = (unsigned)(((N_NODES + 15) / 16 + 3) / 4);  // wave = 16 rows

  // ---- layer 1: xw1 = x @ W1 (bf16 WMMA, f32 accum) ----
  cvt_f32_bf16<<<nb((long long)N_NODES * F_IN, TB), TB, 0, stream>>>(x, xbf, (long long)N_NODES * F_IN);
  pack_b_bf16<<<nb((long long)F_IN * 64, TB), TB, 0, stream>>>(W1, w1p, F_IN, 64);
  gemm_bf16_wmma<<<dim3(64 / 64, mBlocks), 128, 0, stream>>>(xbf, w1p, xw1, N_NODES, F_IN, 64);

  fill_f32<<<nb((long long)N_NODES * HEADS, TB), TB, 0, stream>>>(amax, -__builtin_inff(), (long long)N_NODES * HEADS);
  fill_f32<<<nb((long long)N_NODES * HEADS, TB), TB, 0, stream>>>(den, 0.f, (long long)N_NODES * HEADS);
  fill_f32<<<nb((long long)N_NODES * 64, TB), TB, 0, stream>>>(agg1, 0.f, (long long)N_NODES * 64);

  edge_alpha<C1><<<nb((long long)E2 * HEADS, TB), TB, 0, stream>>>(xw1, esrc, edst, attl1, attr1, alp, amax);
  edge_expsum<<<nb((long long)E2 * HEADS, TB), TB, 0, stream>>>(edst, alp, amax, den);
  edge_agg<C1><<<nb((long long)E2 * HEADS * (C1 / 4), TB), TB, 0, stream>>>(xw1, esrc, edst, alp, den, agg1);
  finalize1<<<nb((long long)N_NODES * 64, TB), TB, 0, stream>>>(agg1, b1, hbf);

  // ---- layer 2: xw2 = h @ W2 ----
  pack_b_bf16<<<nb((long long)64 * 320, TB), TB, 0, stream>>>(W2, w2p, 64, 320);
  gemm_bf16_wmma<<<dim3(320 / 64, mBlocks), 128, 0, stream>>>(hbf, w2p, xw2, N_NODES, 64, 320);

  fill_f32<<<nb((long long)N_NODES * HEADS, TB), TB, 0, stream>>>(amax, -__builtin_inff(), (long long)N_NODES * HEADS);
  fill_f32<<<nb((long long)N_NODES * HEADS, TB), TB, 0, stream>>>(den, 0.f, (long long)N_NODES * HEADS);
  fill_f32<<<nb((long long)N_NODES * HEADS * N_CLASSES, TB), TB, 0, stream>>>(agg2, 0.f, (long long)N_NODES * HEADS * N_CLASSES);

  edge_alpha<N_CLASSES><<<nb((long long)E2 * HEADS, TB), TB, 0, stream>>>(xw2, esrc, edst, attl2, attr2, alp, amax);
  edge_expsum<<<nb((long long)E2 * HEADS, TB), TB, 0, stream>>>(edst, alp, amax, den);
  edge_agg<N_CLASSES><<<nb((long long)E2 * HEADS * (N_CLASSES / 4), TB), TB, 0, stream>>>(xw2, esrc, edst, alp, den, agg2);

  finalize2<<<nb((long long)N_NODES, TB), TB, 0, stream>>>(agg2, b2, out);
}
